// CapsuleNetwork_31490700214847
// MI455X (gfx1250) — matrix operand order repaired and checked by reading, not compiled, not run on hardware
//
#include <hip/hip_runtime.h>
#include <hip/hip_bf16.h>
#include <math.h>

// ---------------- problem constants ----------------
#define B_SZ   32
#define HIMG   36
#define H1     28          // after conv1 (VALID 9x9)
#define H2     20          // after primary conv
#define COUT1  256
#define KSZ    9
#define KK81   81
#define NPU    8
#define CUOC   32
#define NOUT   10
#define OUS    16
#define P_TOT  12800       // CUOC*H2*H2
#define M_TOT  12800       // B*H2*H2
#define N_OC   256
#define NSTEP  (KK81 * 8)  // 648 K-steps of 32

typedef __attribute__((ext_vector_type(16))) __bf16 v16bf;
typedef __attribute__((ext_vector_type(8)))  float  v8f;

__device__ __forceinline__ unsigned short f2bf(float f) {
  unsigned int u = __float_as_uint(f);
  u += 0x7FFFu + ((u >> 16) & 1u);          // round-to-nearest-even
  return (unsigned short)(u >> 16);
}

// build a 16xbf16 WMMA fragment from two 16B global loads
__device__ __forceinline__ v16bf ld_pair(const unsigned short* p0, const unsigned short* p1) {
  union { uint4 q[2]; v16bf v; } t;
  t.q[0] = *reinterpret_cast<const uint4*>(p0);
  t.q[1] = *reinterpret_cast<const uint4*>(p1);
  return t.v;
}

__device__ __forceinline__ v8f wmma_bf16(v16bf a, v16bf b, v8f c) {
  return __builtin_amdgcn_wmma_f32_16x16x32_bf16(false, a, false, b, (short)0, c, false, false);
}

// ---------------- K0: conv1 + ReLU -> bf16 NHWC ----------------
// one thread per output element; t = ((b*28+y)*28+x)*256 + oc  (matches NHWC store)
__global__ void caps_conv1(const float* __restrict__ x, const float* __restrict__ w,
                           const float* __restrict__ bias, unsigned short* __restrict__ h) {
  int t = blockIdx.x * blockDim.x + threadIdx.x;
  int oc   = t & 255;
  int rest = t >> 8;
  int xx = rest % H1; rest /= H1;
  int y  = rest % H1;
  int b  = rest / H1;
  const float* img = x + b * HIMG * HIMG;
  const float* wp  = w + oc * (KSZ * KSZ);
  float acc = bias[oc];
  #pragma unroll
  for (int ky = 0; ky < KSZ; ++ky)
    #pragma unroll
    for (int kx = 0; kx < KSZ; ++kx)
      acc += img[(y + ky) * HIMG + (xx + kx)] * wp[ky * KSZ + kx];
  acc = fmaxf(acc, 0.0f);
  h[t] = f2bf(acc);
}

// ---------------- K1: primary weights -> WMMA-swizzled bf16 B ----------------
// Bsw[step(kk*8+c)][half][n][e] with cin = c*32 + (e&8)*2 + half*8 + (e&7)
__global__ void caps_prep_bsw(const float* __restrict__ pw, unsigned short* __restrict__ bsw) {
  int f = blockIdx.x * 256 + threadIdx.x;        // 81*8*2*256*16 = 5,308,416 exact
  int e  = f & 15;
  int n  = (f >> 4) & 255;
  int hh = (f >> 12) & 1;
  int c  = (f >> 13) & 7;
  int kk = f >> 16;                               // 0..80
  int cin = c * 32 + (e & 8) * 2 + hh * 8 + (e & 7);
  bsw[f] = f2bf(pw[(n * COUT1 + cin) * KK81 + kk]);
}

// ---------------- K2: primary conv as implicit GEMM with WMMA ----------------
// one wave per 32M x 64N tile; K flattened to 648 steps of 32 (81 shifts x 8 cin-chunks)
// double-buffered in registers: stage k+1's 12 b128 loads overlap stage k's 8 WMMAs
__global__ __launch_bounds__(32) void caps_primary_wmma(
    const unsigned short* __restrict__ h, const unsigned short* __restrict__ bsw,
    const float* __restrict__ pbias, float* __restrict__ u) {
  int lane = threadIdx.x;
  int lm   = lane & 15;
  int hsel = (lane < 16) ? 0 : 1;
  int m0 = blockIdx.x * 32;
  int n0 = blockIdx.y * 64;

  // per-lane im2col row bases (NHWC element offsets)
  int row0 = m0 + lm;
  int row1 = row0 + 16;
  int b0 = row0 / 400, r0 = row0 % 400;
  int b1 = row1 / 400, r1 = row1 % 400;
  int base0 = ((b0 * H1 + (r0 / H2)) * H1 + (r0 % H2)) * 256;
  int base1 = ((b1 * H1 + (r1 / H2)) * H1 + (r1 % H2)) * 256;
  // ISA 16-bit A layout: lanes<16 take K {0..7,16..23}, lanes>=16 take {8..15,24..31}
  int aoff = hsel ? 8 : 0;

  v8f acc[2][4];
  #pragma unroll
  for (int i = 0; i < 2; ++i)
    #pragma unroll
    for (int j = 0; j < 4; ++j) {
      v8f z = {0.f,0.f,0.f,0.f,0.f,0.f,0.f,0.f};
      acc[i][j] = z;
    }

  v16bf Ab[2][2];
  v16bf Bb[2][4];

  auto load_stage = [&](int step, int st) {
    int kk = step >> 3, c = step & 7;
    int koff = ((kk / 9) * H1 + (kk % 9)) * 256 + c * 32 + aoff;
    const unsigned short* p0 = h + base0 + koff;
    const unsigned short* p1 = h + base1 + koff;
    Ab[st][0] = ld_pair(p0, p0 + 16);
    Ab[st][1] = ld_pair(p1, p1 + 16);
    const unsigned short* bp =
        bsw + (size_t)((step * 2 + hsel) * 256 + n0 + lm) * 16;
    Bb[st][0] = ld_pair(bp,       bp + 8);
    Bb[st][1] = ld_pair(bp + 256, bp + 264);
    Bb[st][2] = ld_pair(bp + 512, bp + 520);
    Bb[st][3] = ld_pair(bp + 768, bp + 776);
  };

  auto compute_stage = [&](int st) {
    #pragma unroll
    for (int j = 0; j < 4; ++j) {
      acc[0][j] = wmma_bf16(Ab[st][0], Bb[st][j], acc[0][j]);
      acc[1][j] = wmma_bf16(Ab[st][1], Bb[st][j], acc[1][j]);
    }
  };

  load_stage(0, 0);
  for (int step = 0; step < NSTEP; step += 2) {   // NSTEP = 648, even
    load_stage(step + 1, 1);                      // prefetch while computing stage 0
    compute_stage(0);
    if (step + 2 < NSTEP) load_stage(step + 2, 0);
    compute_stage(1);
  }

  // D layout: VGPR r -> M = r + (hsel?8:0); N = lane&15
  #pragma unroll
  for (int i = 0; i < 2; ++i)
    #pragma unroll
    for (int j = 0; j < 4; ++j) {
      int col = n0 + j * 16 + lm;
      float bb = pbias[col];
      int rbase = m0 + i * 16 + hsel * 8;
      #pragma unroll
      for (int r = 0; r < 8; ++r)
        u[(size_t)(rbase + r) * 256 + col] = acc[i][j][r] + bb;
    }
}

// ---------------- K3a: squash magnitudes over P per (b,unit) ----------------
__global__ void caps_sq_mag(const float* __restrict__ u, float* __restrict__ magsq) {
  __shared__ float red[256];
  int b = blockIdx.x >> 3, unit = blockIdx.x & 7;
  int tid = threadIdx.x;
  float s = 0.f;
  for (int p = tid; p < P_TOT; p += 256) {
    int ch = p / 400, pix = p % 400;
    float v = u[(size_t)(b * 400 + pix) * 256 + unit * 32 + ch];
    s += v * v;
  }
  red[tid] = s; __syncthreads();
  for (int st = 128; st > 0; st >>= 1) {
    if (tid < st) red[tid] += red[tid + st];
    __syncthreads();
  }
  if (tid == 0) magsq[blockIdx.x] = red[0];
}

// ---------------- K3b: apply squash, emit xp[b][p][u] ----------------
__global__ void caps_sq_scale(const float* __restrict__ u, const float* __restrict__ magsq,
                              float* __restrict__ xp) {
  int t = blockIdx.x * 256 + threadIdx.x;     // B*P = 409600 exact
  int b = t / P_TOT, p = t % P_TOT;
  int ch = p / 400, pix = p % 400;
  #pragma unroll
  for (int unit = 0; unit < 8; ++unit) {
    float ms = magsq[b * 8 + unit];
    float sc = ms / ((1.0f + ms) * sqrtf(ms));
    xp[(size_t)t * 8 + unit] =
        u[(size_t)(b * 400 + pix) * 256 + unit * 32 + ch] * sc;
  }
}

// ---------------- routing ----------------
__global__ void caps_zero(float* __restrict__ bij) {
  bij[blockIdx.x * 256 + threadIdx.x] = 0.0f;
}

__global__ void caps_softmax_p(const float* __restrict__ bij, float* __restrict__ cc) {
  __shared__ float red[256];
  __shared__ float s_mx, s_sum;
  int o = blockIdx.x, tid = threadIdx.x;
  float mx = -1e30f;
  for (int p = tid; p < P_TOT; p += 256) mx = fmaxf(mx, bij[p * NOUT + o]);
  red[tid] = mx; __syncthreads();
  for (int st = 128; st > 0; st >>= 1) {
    if (tid < st) red[tid] = fmaxf(red[tid], red[tid + st]);
    __syncthreads();
  }
  if (tid == 0) s_mx = red[0];
  __syncthreads();
  mx = s_mx;
  float sum = 0.f;
  for (int p = tid; p < P_TOT; p += 256) sum += __expf(bij[p * NOUT + o] - mx);
  red[tid] = sum; __syncthreads();
  for (int st = 128; st > 0; st >>= 1) {
    if (tid < st) red[tid] += red[tid + st];
    __syncthreads();
  }
  if (tid == 0) s_sum = red[0];
  __syncthreads();
  float inv = 1.0f / s_sum;
  for (int p = tid; p < P_TOT; p += 256)
    cc[p * NOUT + o] = __expf(bij[p * NOUT + o] - mx) * inv;
}

// s_j[b,o,s] = sum_p c[p,o] * (sum_u W[p,o,s,u]*xp[b,p,u]); one block per (b,o)
__global__ void caps_compute_s(const float* __restrict__ cc, const float* __restrict__ xp,
                               const float* __restrict__ Wd, float* __restrict__ sj) {
  __shared__ float red[256];
  int b = blockIdx.x / NOUT, o = blockIdx.x % NOUT;
  int tid = threadIdx.x;
  float acc[OUS];
  #pragma unroll
  for (int s = 0; s < OUS; ++s) acc[s] = 0.f;
  for (int p = tid; p < P_TOT; p += 256) {
    float cpo = cc[p * NOUT + o];
    const float* xv = xp + (size_t)(b * P_TOT + p) * 8;
    float x0=xv[0],x1=xv[1],x2=xv[2],x3=xv[3],x4=xv[4],x5=xv[5],x6=xv[6],x7=xv[7];
    const float* wp = Wd + (size_t)(p * NOUT + o) * (OUS * NPU);
    #pragma unroll
    for (int s = 0; s < OUS; ++s) {
      const float* ws8 = wp + s * 8;
      float uh = ws8[0]*x0 + ws8[1]*x1 + ws8[2]*x2 + ws8[3]*x3 +
                 ws8[4]*x4 + ws8[5]*x5 + ws8[6]*x6 + ws8[7]*x7;
      acc[s] += cpo * uh;
    }
  }
  for (int s = 0; s < OUS; ++s) {
    red[tid] = acc[s]; __syncthreads();
    for (int st = 128; st > 0; st >>= 1) {
      if (tid < st) red[tid] += red[tid + st];
      __syncthreads();
    }
    if (tid == 0) sj[(b * NOUT + o) * OUS + s] = red[0];
    __syncthreads();
  }
}

// squash over the O axis; optionally also write d_out (final iteration)
__global__ void caps_squash_v(const float* __restrict__ sj, float* __restrict__ v,
                              float* __restrict__ out) {
  int t = blockIdx.x * 256 + threadIdx.x;   // 512 = B*OUS
  if (t >= B_SZ * OUS) return;
  int b = t >> 4, s = t & 15;
  float ms = 0.f;
  #pragma unroll
  for (int o = 0; o < NOUT; ++o) {
    float x = sj[(b * NOUT + o) * OUS + s];
    ms += x * x;
  }
  float sc = ms / ((1.0f + ms) * sqrtf(ms));
  #pragma unroll
  for (int o = 0; o < NOUT; ++o) {
    int idx = (b * NOUT + o) * OUS + s;
    float val = sj[idx] * sc;
    v[idx] = val;
    if (out) out[idx] = val;
  }
}

// b_ij[p,o] += mean_b sum_s u_hat(b,p,o,s)*v[b,o,s]; W held in regs in two s-halves
__global__ void caps_update_b(const float* __restrict__ xp, const float* __restrict__ Wd,
                              const float* __restrict__ v, float* __restrict__ bij) {
  int t = blockIdx.x * 256 + threadIdx.x;   // P*NOUT = 128000 exact
  int p = t / NOUT, o = t % NOUT;
  const float* wp = Wd + (size_t)(p * NOUT + o) * (OUS * NPU);
  float agree = 0.f;
  #pragma unroll
  for (int sh = 0; sh < 2; ++sh) {
    float w[64];
    #pragma unroll
    for (int i = 0; i < 64; ++i) w[i] = wp[sh * 64 + i];
    for (int b = 0; b < B_SZ; ++b) {
      const float* xv = xp + (size_t)(b * P_TOT + p) * 8;
      float x0=xv[0],x1=xv[1],x2=xv[2],x3=xv[3],x4=xv[4],x5=xv[5],x6=xv[6],x7=xv[7];
      #pragma unroll
      for (int s = 0; s < 8; ++s) {
        const float* w8 = &w[s * 8];
        float uh = w8[0]*x0 + w8[1]*x1 + w8[2]*x2 + w8[3]*x3 +
                   w8[4]*x4 + w8[5]*x5 + w8[6]*x6 + w8[7]*x7;
        agree += uh * v[(b * NOUT + o) * OUS + sh * 8 + s];
      }
    }
  }
  bij[t] += agree * (1.0f / (float)B_SZ);
}

// ---------------- host side ----------------
extern "C" void kernel_launch(void* const* d_in, const int* in_sizes, int n_in,
                              void* d_out, int out_size, void* d_ws, size_t ws_size,
                              hipStream_t stream) {
  const float* x   = (const float*)d_in[0];
  const float* c1w = (const float*)d_in[1];
  const float* c1b = (const float*)d_in[2];
  const float* pw  = (const float*)d_in[3];
  const float* pb  = (const float*)d_in[4];
  const float* Wd  = (const float*)d_in[5];
  float* out = (float*)d_out;

  char* base = (char*)d_ws;
  size_t off = 0;
  auto alloc = [&](size_t bytes) -> void* {
    void* p = base + off;
    off += (bytes + 255) & ~(size_t)255;
    return p;
  };
  unsigned short* h_bf  = (unsigned short*)alloc((size_t)B_SZ * H1 * H1 * 256 * 2);
  unsigned short* bsw   = (unsigned short*)alloc((size_t)NSTEP * 2 * 256 * 16 * 2);
  float* u     = (float*)alloc((size_t)M_TOT * 256 * 4);
  float* magsq = (float*)alloc((size_t)B_SZ * NPU * 4);
  float* xp    = (float*)alloc((size_t)B_SZ * P_TOT * 8 * 4);
  float* bij   = (float*)alloc((size_t)P_TOT * NOUT * 4);
  float* cc    = (float*)alloc((size_t)P_TOT * NOUT * 4);
  float* sj    = (float*)alloc((size_t)B_SZ * NOUT * OUS * 4);
  float* vbuf  = (float*)alloc((size_t)B_SZ * NOUT * OUS * 4);

  // conv1 + relu -> bf16 NHWC
  caps_conv1<<<B_SZ * H1 * H1, 256, 0, stream>>>(x, c1w, c1b, h_bf);
  // weight swizzle for WMMA B fragments
  caps_prep_bsw<<<(NSTEP * 2 * 256 * 16) / 256, 256, 0, stream>>>(pw, bsw);
  // primary conv: implicit GEMM, WMMA bf16, register double-buffered
  caps_primary_wmma<<<dim3(M_TOT / 32, N_OC / 64), 32, 0, stream>>>(h_bf, bsw, pb, u);
  // squash over P
  caps_sq_mag<<<B_SZ * NPU, 256, 0, stream>>>(u, magsq);
  caps_sq_scale<<<(B_SZ * P_TOT) / 256, 256, 0, stream>>>(u, magsq, xp);
  // routing
  caps_zero<<<(P_TOT * NOUT) / 256, 256, 0, stream>>>(bij);
  for (int it = 0; it < 3; ++it) {
    caps_softmax_p<<<NOUT, 256, 0, stream>>>(bij, cc);
    caps_compute_s<<<B_SZ * NOUT, 256, 0, stream>>>(cc, xp, Wd, sj);
    caps_squash_v<<<2, 256, 0, stream>>>(sj, vbuf, (it == 2) ? out : (float*)nullptr);
    if (it < 2)
      caps_update_b<<<(P_TOT * NOUT) / 256, 256, 0, stream>>>(xp, Wd, vbuf, bij);
  }
}
